// SlimMambaBlock_71777493451117
// MI455X (gfx1250) — matrix operand order repaired and checked by reading, not compiled, not run on hardware
//
#include <hip/hip_runtime.h>

typedef __attribute__((ext_vector_type(8)))  __bf16 v8bf;
typedef __attribute__((ext_vector_type(16))) __bf16 v16bf;
typedef __attribute__((ext_vector_type(8)))  float  v8f;
typedef __attribute__((ext_vector_type(4)))  float  v4f;

#define D_MODEL 256
#define E_INNER 512
#define N1      1024
#define SEQ_K   4096
#define BATCH   8
#define M_TOK   (BATCH*SEQ_K)   // 32768

#define NCHUNK  16
#define CH_LEN  (SEQ_K/NCHUNK)  // 256

// fast sigmoid: v_exp_f32 + v_rcp_f32, no IEEE divide sequence
__device__ __forceinline__ float sigf(float x){
  return __builtin_amdgcn_rcpf(1.0f + __expf(-x));
}

// ---------------- f32 -> bf16 weight conversion ----------------
__global__ void k_cvt_bf16(const float* __restrict__ s, __bf16* __restrict__ d, int n){
  int i = blockIdx.x*blockDim.x + threadIdx.x;
  if (i < n) d[i] = (__bf16)s[i];
}

// ---------------- RMSNorm, one wave32 per token, writes bf16 h ----------------
__global__ void k_rms(const float* __restrict__ x, const float* __restrict__ w,
                      __bf16* __restrict__ h){
  int wave = threadIdx.x >> 5;
  int lane = threadIdx.x & 31;
  int tok  = blockIdx.x*8 + wave;
  const float* xp = x + (size_t)tok*D_MODEL + lane*8;
  v4f a = *(const v4f*)xp;
  v4f b = *(const v4f*)(xp+4);
  float ss = a[0]*a[0]+a[1]*a[1]+a[2]*a[2]+a[3]*a[3]
           + b[0]*b[0]+b[1]*b[1]+b[2]*b[2]+b[3]*b[3];
  ss += __shfl_xor(ss, 16, 32);
  ss += __shfl_xor(ss,  8, 32);
  ss += __shfl_xor(ss,  4, 32);
  ss += __shfl_xor(ss,  2, 32);
  ss += __shfl_xor(ss,  1, 32);
  float r = rsqrtf(ss*(1.0f/(float)D_MODEL) + 1e-5f);
  const float* wp = w + lane*8;
  v8bf o;
  #pragma unroll
  for (int i=0;i<4;i++) o[i]   = (__bf16)(a[i]*r*wp[i]);
  #pragma unroll
  for (int i=0;i<4;i++) o[4+i] = (__bf16)(b[i]*r*wp[4+i]);
  *(v8bf*)(h + (size_t)tok*D_MODEL + lane*8) = o;
}

// ---------------- WMMA bf16 GEMM: C[M,N] = A[M,Kd] * B[N,Kd]^T ----------------
// Block tile: 128(M) x 128(N) x 64(K); 8 waves in a 4x2 grid; wave tile 32x64
// = 2x4 WMMA 16x16 tiles. Double-buffered LDS (global->reg of tile t+1 overlaps
// WMMA on tile t; one barrier per K-tile).
#define BM 128
#define BN 128
#define BK 64
#define SA 72   // padded LDS row stride (bf16), multiple of 8 -> aligned b128, no bank conflicts
#define SB 72

// MODE 0: uv = A*B^T, silu, split -> u16 (n<512) / g16 (n>=512), bf16
// MODE 1: lam = sigmoid(A*B^T + bias) -> f32
// MODE 2: out = xres + A*B^T -> f32
template<int MODE>
__global__ __launch_bounds__(256)
void k_gemm(const __bf16* __restrict__ A, const __bf16* __restrict__ B, int Kd,
            const float* __restrict__ bias, const float* __restrict__ xres,
            float* __restrict__ outf, __bf16* __restrict__ out_u,
            __bf16* __restrict__ out_g)
{
  __shared__ __align__(16) __bf16 As[2*BM*SA];
  __shared__ __align__(16) __bf16 Bs[2*BN*SB];
  const int tid  = threadIdx.x;
  const int lane = tid & 31, wave = tid >> 5;
  const int wm = wave & 3, wn = wave >> 2;      // 4(M) x 2(N) wave grid
  const int lm = lane & 15, lh = lane >> 4;
  const int m0 = blockIdx.x * BM;
  const int n0 = blockIdx.y * BN;

  // staging coordinates: 128x64 bf16 tile = 1024 v8bf chunks, 4 per thread
  const int sRow = tid >> 3;            // base row (stride 32 rows per i)
  const int sCol = (tid & 7) << 3;

  v8f acc[2][4];
  {
    v8f vz = {0.f,0.f,0.f,0.f,0.f,0.f,0.f,0.f};
    #pragma unroll
    for (int i=0;i<2;i++)
      #pragma unroll
      for (int j=0;j<4;j++) acc[i][j] = vz;
  }

  const int ntile = Kd / BK;

  // prologue: stage tile 0 into buffer 0
  #pragma unroll
  for (int i=0;i<4;i++){
    int row = sRow + i*32;
    *(v8bf*)&As[row*SA + sCol] = *(const v8bf*)&A[(size_t)(m0+row)*Kd + sCol];
    *(v8bf*)&Bs[row*SB + sCol] = *(const v8bf*)&B[(size_t)(n0+row)*Kd + sCol];
  }
  __syncthreads();

  for (int t = 0; t < ntile; t++){
    // issue global loads for tile t+1 early (overlap with WMMA below)
    v8bf Areg[4], Breg[4];
    if (t+1 < ntile){
      int kt = (t+1)*BK;
      #pragma unroll
      for (int i=0;i<4;i++){
        int row = sRow + i*32;
        Areg[i] = *(const v8bf*)&A[(size_t)(m0+row)*Kd + kt + sCol];
        Breg[i] = *(const v8bf*)&B[(size_t)(n0+row)*Kd + kt + sCol];
      }
    }

    const __bf16* Ab = As + (t&1)*(BM*SA);
    const __bf16* Bb = Bs + (t&1)*(BN*SB);

    #pragma unroll
    for (int ks = 0; ks < 2; ks++){      // two 16x16x32 K-steps per LDS tile
      v16bf af[2], bfr[4];
      #pragma unroll
      for (int ti=0;ti<2;ti++){
        // A fragment (16x32): lane lm = row; elems 0..7 = K 8*lh.., 8..15 = K 16+8*lh..
        int r = wm*32 + ti*16 + lm;
        v8bf lo = *(const v8bf*)&Ab[r*SA + ks*32 + lh*8];
        v8bf hi = *(const v8bf*)&Ab[r*SA + ks*32 + 16 + lh*8];
        #pragma unroll
        for (int i=0;i<8;i++){ af[ti][i]=lo[i]; af[ti][8+i]=hi[i]; }
      }
      #pragma unroll
      for (int tj=0;tj<4;tj++){
        // B fragment (32x16): lane lm = col N; elems 0..7 = K 8*lh.., 8..15 = K 16+8*lh..
        int nn = wn*64 + tj*16 + lm;
        v8bf b0 = *(const v8bf*)&Bb[nn*SB + ks*32 + lh*8];
        v8bf b1 = *(const v8bf*)&Bb[nn*SB + ks*32 + 16 + lh*8];
        #pragma unroll
        for (int i=0;i<8;i++){ bfr[tj][i]=b0[i]; bfr[tj][8+i]=b1[i]; }
      }
      #pragma unroll
      for (int ti=0;ti<2;ti++)
        #pragma unroll
        for (int tj=0;tj<4;tj++)
          acc[ti][tj] = __builtin_amdgcn_wmma_f32_16x16x32_bf16(
              false, af[ti], false, bfr[tj], (short)0, acc[ti][tj], false, false);
    }

    // store tile t+1 into the other buffer
    if (t+1 < ntile){
      __bf16* Aw = As + ((t+1)&1)*(BM*SA);
      __bf16* Bw = Bs + ((t+1)&1)*(BN*SB);
      #pragma unroll
      for (int i=0;i<4;i++){
        int row = sRow + i*32;
        *(v8bf*)&Aw[row*SA + sCol] = Areg[i];
        *(v8bf*)&Bw[row*SB + sCol] = Breg[i];
      }
    }
    __syncthreads();
  }

  // epilogue: C layout per 16x16 tile: lane (lh,lm), VGPR r -> (m = 8*lh + r, n = lm)
  #pragma unroll
  for (int ti=0;ti<2;ti++){
    #pragma unroll
    for (int tj=0;tj<4;tj++){
      int gn = n0 + wn*64 + tj*16 + lm;
      #pragma unroll
      for (int r=0;r<8;r++){
        int gm = m0 + wm*32 + ti*16 + lh*8 + r;
        float v = acc[ti][tj][r];
        if (MODE == 0) {
          float s = v * sigf(v);                       // silu (both halves)
          if (gn < E_INNER) out_u[(size_t)gm*E_INNER + gn]             = (__bf16)s;
          else              out_g[(size_t)gm*E_INNER + (gn - E_INNER)] = (__bf16)s;
        } else if (MODE == 1) {
          outf[(size_t)gm*E_INNER + gn] = sigf(v + bias[gn]);
        } else {
          size_t idx = (size_t)gm*D_MODEL + gn;
          outf[idx] = xres[idx] + v;
        }
      }
    }
  }
}

// ---------------- chunked linear-recurrence scan ----------------
// s_k = l_k * s_{k-1} + (1-l_k) * u_k  ; per chunk: s_out = P * s_in + S
__global__ void k_scan1(const float* __restrict__ lam, const __bf16* __restrict__ u,
                        float* __restrict__ Pb, float* __restrict__ Sb){
  int bx = blockIdx.x;                 // 256 blocks: b(8) x chunk(16) x eblk(2)
  int b = bx >> 5, chunk = (bx >> 1) & 15, eblk = bx & 1;
  int e = eblk*256 + threadIdx.x;
  int chain = b*E_INNER + e;
  size_t base = ((size_t)b*SEQ_K + (size_t)chunk*CH_LEN)*E_INNER + e;
  float P = 1.0f, S = 0.0f;
  for (int k=0;k<CH_LEN;k++){
    float l  = lam[base];
    float uu = (float)u[base];
    S = l*S + (1.0f-l)*uu;
    P *= l;
    base += E_INNER;
  }
  Pb[chain*NCHUNK + chunk] = P;
  Sb[chain*NCHUNK + chunk] = S;
}

__global__ void k_scanfix(const float* __restrict__ Pb, const float* __restrict__ Sb,
                          float* __restrict__ st){
  int chain = blockIdx.x*blockDim.x + threadIdx.x;  // 4096 chains
  float s = 0.0f;
  #pragma unroll
  for (int c=0;c<NCHUNK;c++){
    st[chain*NCHUNK + c] = s;                       // incoming state for chunk c
    s = Pb[chain*NCHUNK + c]*s + Sb[chain*NCHUNK + c];
  }
}

__global__ void k_scan2(const float* __restrict__ lam, const __bf16* __restrict__ u,
                        const __bf16* __restrict__ g, const float* __restrict__ st,
                        __bf16* __restrict__ p){
  int bx = blockIdx.x;
  int b = bx >> 5, chunk = (bx >> 1) & 15, eblk = bx & 1;
  int e = eblk*256 + threadIdx.x;
  int chain = b*E_INNER + e;
  size_t base = ((size_t)b*SEQ_K + (size_t)chunk*CH_LEN)*E_INNER + e;
  float s = st[chain*NCHUNK + chunk];
  for (int k=0;k<CH_LEN;k++){
    float l  = lam[base];
    float uu = (float)u[base];
    s = l*s + (1.0f-l)*uu;
    float gg = (float)g[base];
    p[base] = (__bf16)(s*gg);                       // fused s * silu(z)
    base += E_INNER;
  }
}

// ---------------- launch ----------------
extern "C" void kernel_launch(void* const* d_in, const int* in_sizes, int n_in,
                              void* d_out, int out_size, void* d_ws, size_t ws_size,
                              hipStream_t stream){
  const float* x    = (const float*)d_in[0];
  const float* nw   = (const float*)d_in[1];
  const float* Win  = (const float*)d_in[2];
  const float* Wdt  = (const float*)d_in[3];
  const float* bdt  = (const float*)d_in[4];
  const float* Wout = (const float*)d_in[5];
  float* out = (float*)d_out;

  char* ws = (char*)d_ws;
  auto alloc = [&](size_t bytes)->char*{
    char* p = ws; ws += (bytes + 255) & ~(size_t)255; return p;
  };
  __bf16* h16    = (__bf16*)alloc((size_t)M_TOK*D_MODEL*2);      // 16 MB
  __bf16* Win16  = (__bf16*)alloc((size_t)N1*D_MODEL*2);
  __bf16* Wdt16  = (__bf16*)alloc((size_t)E_INNER*E_INNER*2);
  __bf16* Wout16 = (__bf16*)alloc((size_t)D_MODEL*E_INNER*2);
  __bf16* u16    = (__bf16*)alloc((size_t)M_TOK*E_INNER*2);      // 32 MB
  __bf16* g16    = (__bf16*)alloc((size_t)M_TOK*E_INNER*2);      // 32 MB
  float*  lam    = (float*) alloc((size_t)M_TOK*E_INNER*4);      // 64 MB
  __bf16* p16    = (__bf16*)alloc((size_t)M_TOK*E_INNER*2);      // 32 MB
  float*  Pb     = (float*) alloc((size_t)BATCH*E_INNER*NCHUNK*4);
  float*  Sb     = (float*) alloc((size_t)BATCH*E_INNER*NCHUNK*4);
  float*  st     = (float*) alloc((size_t)BATCH*E_INNER*NCHUNK*4);

  k_cvt_bf16<<<(N1*D_MODEL+255)/256, 256, 0, stream>>>(Win, Win16, N1*D_MODEL);
  k_cvt_bf16<<<(E_INNER*E_INNER+255)/256, 256, 0, stream>>>(Wdt, Wdt16, E_INNER*E_INNER);
  k_cvt_bf16<<<(D_MODEL*E_INNER+255)/256, 256, 0, stream>>>(Wout, Wout16, D_MODEL*E_INNER);

  k_rms<<<M_TOK/8, 256, 0, stream>>>(x, nw, h16);

  k_gemm<0><<<dim3(M_TOK/BM, N1/BN), 256, 0, stream>>>(
      h16, Win16, D_MODEL, nullptr, nullptr, nullptr, u16, g16);
  k_gemm<1><<<dim3(M_TOK/BM, E_INNER/BN), 256, 0, stream>>>(
      u16, Wdt16, E_INNER, bdt, nullptr, lam, nullptr, nullptr);

  k_scan1<<<256, 256, 0, stream>>>(lam, u16, Pb, Sb);
  k_scanfix<<<16, 256, 0, stream>>>(Pb, Sb, st);
  k_scan2<<<256, 256, 0, stream>>>(lam, u16, g16, st, p16);

  k_gemm<2><<<dim3(M_TOK/BM, D_MODEL/BN), 256, 0, stream>>>(
      p16, Wout16, E_INNER, nullptr, x, out, nullptr, nullptr);
}
